// TopKSAE_27135603376479
// MI455X (gfx1250) — matrix operand order repaired and checked by reading, not compiled, not run on hardware
//
#include <hip/hip_runtime.h>
#include <hip/hip_bf16.h>
#include <stdint.h>

#define IDIM   1024
#define HDIM   16384
#define NBATCH 8192
#define TOPK   64

typedef __attribute__((ext_vector_type(16))) __bf16 v16bf;
typedef __attribute__((ext_vector_type(8)))  float  v8f;

// ---------------------------------------------------------------------------
// f32 -> (bf16 hi, bf16 lo) split, RNE. a ~= hi+lo to ~16 mantissa bits:
// ah*bh + ah*bl + al*bh on the bf16 WMMA pipe gives ~fp32-accurate GEMM at
// 8x the K-depth/instruction of V_WMMA_F32_16X16X4_F32.
// ---------------------------------------------------------------------------
__device__ __forceinline__ void f32_split_bf16(float f, unsigned short& h, unsigned short& l) {
    unsigned u  = __float_as_uint(f);
    unsigned r  = (u + 0x7FFFu + ((u >> 16) & 1u)) >> 16;
    h = (unsigned short)r;
    float hf = __uint_as_float(r << 16);
    float lo = f - hf;
    unsigned u2 = __float_as_uint(lo);
    unsigned r2 = (u2 + 0x7FFFu + ((u2 >> 16) & 1u)) >> 16;
    l = (unsigned short)r2;
}

// ---------------------------------------------------------------------------
// Kernel 0: one-shot f32 -> bf16 hi/lo plane conversion (x and W_enc).
// Removes ALL conversion VALU work from the GEMM hot loop (x would otherwise
// be re-converted 128x, W_enc 64x).
// ---------------------------------------------------------------------------
__global__ __launch_bounds__(256) void k_bf16split(const float* __restrict__ src,
                                                   unsigned short* __restrict__ hi,
                                                   unsigned short* __restrict__ lo) {
    const int i = blockIdx.x * 256 + threadIdx.x;   // one float4 per thread
    float4 v = ((const float4*)src)[i];
    unsigned short h0,h1,h2,h3,l0,l1,l2,l3;
    f32_split_bf16(v.x,h0,l0); f32_split_bf16(v.y,h1,l1);
    f32_split_bf16(v.z,h2,l2); f32_split_bf16(v.w,h3,l3);
    uint2 hv; hv.x = (unsigned)h0 | ((unsigned)h1 << 16); hv.y = (unsigned)h2 | ((unsigned)h3 << 16);
    uint2 lv; lv.x = (unsigned)l0 | ((unsigned)l1 << 16); lv.y = (unsigned)l2 | ((unsigned)l3 << 16);
    ((uint2*)hi)[i] = hv;
    ((uint2*)lo)[i] = lv;
}

// ---------------------------------------------------------------------------
// Kernel 1: transpose W_dec [IDIM][HDIM] -> Wt [HDIM][IDIM] for coalesced
// sparse-decode gathers.
// ---------------------------------------------------------------------------
__global__ __launch_bounds__(256) void k_transpose(const float* __restrict__ Wd,
                                                   float* __restrict__ Wt) {
    __shared__ float tile[32][33];
    const int tx = threadIdx.x, ty = threadIdx.y;
    const int h0 = blockIdx.x * 32, i0 = blockIdx.y * 32;
#pragma unroll
    for (int j = 0; j < 4; ++j)
        tile[ty + j * 8][tx] = Wd[(size_t)(i0 + ty + j * 8) * HDIM + h0 + tx];
    __syncthreads();
#pragma unroll
    for (int j = 0; j < 4; ++j)
        Wt[(size_t)(h0 + ty + j * 8) * IDIM + i0 + tx] = tile[tx][ty + j * 8];
}

// ---------------------------------------------------------------------------
// Kernel 2: encoder GEMM  pre_act = ReLU(x @ W_enc^T + b_enc) -> z region.
// Block tile 128x128, 8 waves in 4(M)x2(N); wave tile 32x64 = 2x4 micro.
// k-block = 32. LDS double-buffered; global loads for tile i+1 are issued
// before consuming tile i (WMMA phase hides the latency); single barrier
// per iteration. 24 WMMA : 24 ds_load_b128 per wave per k-block.
// ---------------------------------------------------------------------------
__global__ __launch_bounds__(256) void k_encode(const unsigned short* __restrict__ xh,
                                                const unsigned short* __restrict__ xl,
                                                const unsigned short* __restrict__ weh,
                                                const unsigned short* __restrict__ wel,
                                                const float* __restrict__ be,
                                                float* __restrict__ z) {
    constexpr int LDA = 40; // bf16-elem row stride (80B): bank-conflict pad, 16B aligned
    __shared__ unsigned short sAh[2][128 * LDA];
    __shared__ unsigned short sAl[2][128 * LDA];
    __shared__ unsigned short sBh[2][128 * LDA];
    __shared__ unsigned short sBl[2][128 * LDA];

    const int tid    = threadIdx.x;
    const int hTile  = blockIdx.x * 128;
    const int bTile  = blockIdx.y * 128;
    const int lane   = tid & 31;
    const int wv     = tid >> 5;
    const int wm     = (wv & 3) << 5;   // 0,32,64,96 (M)
    const int wn     = (wv >> 2) << 6;  // 0,64       (N)
    const int laneLo = lane & 15;
    const int laneHi = lane >> 4;

    v8f acc[2][4];
#pragma unroll
    for (int mi = 0; mi < 2; ++mi)
#pragma unroll
        for (int ni = 0; ni < 4; ++ni) {
            v8f zz = {0.f,0.f,0.f,0.f,0.f,0.f,0.f,0.f};
            acc[mi][ni] = zz;
        }

    uint4 rAh[2], rAl[2], rBh[2], rBl[2];

#define FETCH(KB)                                                              \
    do {                                                                       \
        _Pragma("unroll")                                                      \
        for (int p = 0; p < 2; ++p) {                                          \
            int task = (p << 8) + tid;                                         \
            int row = task >> 2, q = (task & 3) << 3;                          \
            rAh[p] = *(const uint4*)(xh  + (size_t)(bTile + row) * IDIM + (KB) + q); \
            rAl[p] = *(const uint4*)(xl  + (size_t)(bTile + row) * IDIM + (KB) + q); \
            rBh[p] = *(const uint4*)(weh + (size_t)(hTile + row) * IDIM + (KB) + q); \
            rBl[p] = *(const uint4*)(wel + (size_t)(hTile + row) * IDIM + (KB) + q); \
        }                                                                      \
    } while (0)

#define STORE(BUF)                                                             \
    do {                                                                       \
        _Pragma("unroll")                                                      \
        for (int p = 0; p < 2; ++p) {                                          \
            int task = (p << 8) + tid;                                         \
            int row = task >> 2, q = (task & 3) << 3;                          \
            *(uint4*)&sAh[BUF][row * LDA + q] = rAh[p];                        \
            *(uint4*)&sAl[BUF][row * LDA + q] = rAl[p];                        \
            *(uint4*)&sBh[BUF][row * LDA + q] = rBh[p];                        \
            *(uint4*)&sBl[BUF][row * LDA + q] = rBl[p];                        \
        }                                                                      \
    } while (0)

    FETCH(0);
    STORE(0);
    __syncthreads();

    constexpr int NIT = IDIM / 32;      // 32 k-blocks
    for (int it = 0; it < NIT; ++it) {
        const int cur = it & 1;
        if (it + 1 < NIT) FETCH((it + 1) * 32);   // overlap with WMMA below

        union FB { uint4 q[2]; v16bf v; };
        FB ah[2], al[2];
        // A 16x32 layout: lane<16 -> K {0..7,16..23}; lane>=16 -> K {8..15,24..31}
        const int ka = laneHi << 3;
#pragma unroll
        for (int mi = 0; mi < 2; ++mi) {
            int r = wm + (mi << 4) + laneLo;
            ah[mi].q[0] = *(const uint4*)&sAh[cur][r * LDA + ka];
            ah[mi].q[1] = *(const uint4*)&sAh[cur][r * LDA + ka + 16];
            al[mi].q[0] = *(const uint4*)&sAl[cur][r * LDA + ka];
            al[mi].q[1] = *(const uint4*)&sAl[cur][r * LDA + ka + 16];
        }
        // B 32x16 layout: lane<16 -> K {0..7,16..23}; lane>=16 -> K {8..15,24..31}
        const int kB = laneHi << 3;
#pragma unroll
        for (int ni = 0; ni < 4; ++ni) {
            FB bh, bl;
            int r = wn + (ni << 4) + laneLo;
            bh.q[0] = *(const uint4*)&sBh[cur][r * LDA + kB];
            bh.q[1] = *(const uint4*)&sBh[cur][r * LDA + kB + 16];
            bl.q[0] = *(const uint4*)&sBl[cur][r * LDA + kB];
            bl.q[1] = *(const uint4*)&sBl[cur][r * LDA + kB + 16];
#pragma unroll
            for (int mi = 0; mi < 2; ++mi) {
                acc[mi][ni] = __builtin_amdgcn_wmma_f32_16x16x32_bf16(
                    false, ah[mi].v, false, bh.v, (short)0, acc[mi][ni], false, false);
                acc[mi][ni] = __builtin_amdgcn_wmma_f32_16x16x32_bf16(
                    false, ah[mi].v, false, bl.v, (short)0, acc[mi][ni], false, false);
                acc[mi][ni] = __builtin_amdgcn_wmma_f32_16x16x32_bf16(
                    false, al[mi].v, false, bh.v, (short)0, acc[mi][ni], false, false);
            }
        }

        if (it + 1 < NIT) STORE((it + 1) & 1);
        __syncthreads();
    }
#undef FETCH
#undef STORE

    // Epilogue: bias + ReLU, C/D layout: VGPR r -> M = r + 8*laneHi, N = laneLo
#pragma unroll
    for (int mi = 0; mi < 2; ++mi)
#pragma unroll
        for (int ni = 0; ni < 4; ++ni) {
            int col = hTile + wn + (ni << 4) + laneLo;
            float bias = be[col];
#pragma unroll
            for (int r = 0; r < 8; ++r) {
                int row = bTile + wm + (mi << 4) + (laneHi << 3) + r;
                float v = acc[mi][ni][r] + bias;
                z[(size_t)row * HDIM + col] = v > 0.f ? v : 0.f;
            }
        }
}

// ---------------------------------------------------------------------------
// Kernel 3: exact top-K=64 per row, 4-bit MSB radix select in LDS.
// ReLU output >= 0 so uint order == float order. Zeroes non-kept z entries
// in place, emits compact (idx,val) lists.
// ---------------------------------------------------------------------------
__global__ __launch_bounds__(256) void k_topk(float* __restrict__ z,
                                              int*   __restrict__ idxL,
                                              float* __restrict__ valL) {
    __shared__ unsigned sv[HDIM];
    __shared__ int hist[16];
    __shared__ unsigned s_prefix;
    __shared__ int s_kth, s_tie, s_out;

    const int row = blockIdx.x;
    const int tid = threadIdx.x;
    float* __restrict__ zr = z + (size_t)row * HDIM;

#pragma unroll 4
    for (int j = 0; j < HDIM / 256; ++j) {
        int e = tid + j * 256;
        sv[e] = __float_as_uint(zr[e]);
    }
    if (tid == 0) { s_prefix = 0u; s_kth = TOPK; s_tie = 0; s_out = 0; }
    __syncthreads();

    unsigned prefmask = 0u;
    for (int shift = 28; shift >= 0; shift -= 4) {
        if (tid < 16) hist[tid] = 0;
        __syncthreads();
        unsigned pref = s_prefix;
        for (int j = 0; j < HDIM / 256; ++j) {
            unsigned v = sv[tid + j * 256];
            if ((v & prefmask) == pref)
                atomicAdd(&hist[(v >> shift) & 15], 1);
        }
        __syncthreads();
        if (tid == 0) {
            int kth = s_kth, cum = 0, dsel = 0;
            for (int d = 15; d >= 0; --d) {
                int c = hist[d];
                if (cum + c >= kth) { dsel = d; break; }
                cum += c;
            }
            s_prefix = pref | ((unsigned)dsel << shift);
            s_kth    = kth - cum;
        }
        prefmask |= (0xFu << shift);
        __syncthreads();
    }

    const unsigned T = s_prefix;
    const int rem    = s_kth;

    for (int j = 0; j < HDIM / 256; ++j) {
        int e = tid + j * 256;
        unsigned v = sv[e];
        bool keep;
        if (v > T)       keep = true;
        else if (v == T) keep = (atomicAdd(&s_tie, 1) < rem);
        else             keep = false;
        if (keep) {
            int slot = atomicAdd(&s_out, 1);
            idxL[row * TOPK + slot] = e;
            valL[row * TOPK + slot] = __uint_as_float(v);
        } else {
            zr[e] = 0.0f;
        }
    }
}

// ---------------------------------------------------------------------------
// Kernel 4: sparse decode  x_hat = z @ W_dec^T + b_dec via compact lists.
// 64 gathered Wt rows (L2-resident) x 1024 outputs per batch row.
// ---------------------------------------------------------------------------
__global__ __launch_bounds__(256) void k_decode(const int*   __restrict__ idxL,
                                                const float* __restrict__ valL,
                                                const float* __restrict__ Wt,
                                                const float* __restrict__ bd,
                                                float* __restrict__ xhat) {
    __shared__ int   sIdx[TOPK];
    __shared__ float sVal[TOPK];
    const int row = blockIdx.x;
    const int tid = threadIdx.x;
    if (tid < TOPK) {
        sIdx[tid] = idxL[row * TOPK + tid];
        sVal[tid] = valL[row * TOPK + tid];
    }
    __syncthreads();

    const int i0 = tid << 2;
    float4 acc = *(const float4*)(bd + i0);
#pragma unroll 8
    for (int k = 0; k < TOPK; ++k) {
        float s  = sVal[k];
        float4 w = *(const float4*)(Wt + (size_t)sIdx[k] * IDIM + i0);
        acc.x += s * w.x; acc.y += s * w.y; acc.z += s * w.z; acc.w += s * w.w;
    }
    *(float4*)(xhat + (size_t)row * IDIM + i0) = acc;
}

// ---------------------------------------------------------------------------
extern "C" void kernel_launch(void* const* d_in, const int* in_sizes, int n_in,
                              void* d_out, int out_size, void* d_ws, size_t ws_size,
                              hipStream_t stream) {
    const float* x  = (const float*)d_in[0];
    const float* We = (const float*)d_in[1];
    const float* be = (const float*)d_in[2];
    const float* Wd = (const float*)d_in[3];
    const float* bd = (const float*)d_in[4];

    float* xhat = (float*)d_out;
    float* z    = (float*)d_out + (size_t)NBATCH * IDIM;

    // workspace layout
    char* ws = (char*)d_ws;
    size_t off = 0;
    float* Wt = (float*)(ws + off);                 off += (size_t)HDIM * IDIM * 4;   // 64 MB
    unsigned short* xhp  = (unsigned short*)(ws + off); off += (size_t)NBATCH * IDIM * 2; // 16 MB
    unsigned short* xlp  = (unsigned short*)(ws + off); off += (size_t)NBATCH * IDIM * 2; // 16 MB
    unsigned short* wehp = (unsigned short*)(ws + off); off += (size_t)HDIM * IDIM * 2;   // 32 MB
    unsigned short* welp = (unsigned short*)(ws + off); off += (size_t)HDIM * IDIM * 2;   // 32 MB
    int*   idxL = (int*)(ws + off);                 off += (size_t)NBATCH * TOPK * 4;  // 2 MB
    float* valL = (float*)(ws + off);

    k_bf16split<<<(NBATCH * IDIM / 4) / 256, 256, 0, stream>>>(x, xhp, xlp);
    k_bf16split<<<(HDIM * IDIM / 4) / 256, 256, 0, stream>>>(We, wehp, welp);
    k_transpose<<<dim3(HDIM / 32, IDIM / 32), dim3(32, 8), 0, stream>>>(Wd, Wt);
    k_encode<<<dim3(HDIM / 128, NBATCH / 128), 256, 0, stream>>>(xhp, xlp, wehp, welp, be, z);
    k_topk<<<NBATCH, 256, 0, stream>>>(z, idxL, valL);
    k_decode<<<NBATCH, 256, 0, stream>>>(idxL, valL, Wt, bd, xhat);
}
